// VirtualBlockOutDecoder_62594853372286
// MI455X (gfx1250) — matrix operand order repaired and checked by reading, not compiled, not run on hardware
//
#include <hip/hip_runtime.h>

#define BD 4
#define ND 512
#define NCH 128
#define ECH 32
#define OCH 64
#define NT 32            // ND/16
#define NPAIR 528        // NT*(NT+1)/2
#define EPAD 66          // padded inner dim of epre (bank-conflict break)

typedef __attribute__((ext_vector_type(16))) __bf16 v16bf;
typedef __attribute__((ext_vector_type(8)))  float  v8f;

__device__ __forceinline__ float silu_f(float x) { return x / (1.f + __expf(-x)); }

// ---------------------------------------------------------------------------
// k_prep: W2 = W_e1 @ W_f2  (32x64),  b2 = b_e1 @ W_f2 + b_f2  (64)
// ---------------------------------------------------------------------------
__global__ __launch_bounds__(256) void k_prep(
    const float* __restrict__ W_e1, const float* __restrict__ b_e1,
    const float* __restrict__ W_f2, const float* __restrict__ b_f2,
    float* __restrict__ W2, float* __restrict__ b2)
{
  for (int idx = threadIdx.x; idx < 32 * 64; idx += 256) {
    int k = idx >> 6, c = idx & 63;
    float a = 0.f;
    for (int m = 0; m < 64; m++) a += W_e1[k * 64 + m] * W_f2[m * 64 + c];
    W2[idx] = a;
  }
  for (int c = threadIdx.x; c < 64; c += 256) {
    float a = b_f2[c];
    for (int m = 0; m < 64; m++) a += b_e1[m] * W_f2[m * 64 + c];
    b2[c] = a;
  }
}

// ---------------------------------------------------------------------------
// k_reduce: masked in/out degree-normalized reductions of edge -> x_in, x_out
// one block per (b,i); 8 j-groups x 32 channel lanes. edge fits in L2 (128MB
// < 192MB), so the column pass and later k_edge re-read are L2 hits.
// ---------------------------------------------------------------------------
__global__ __launch_bounds__(256) void k_reduce(
    const float* __restrict__ edge, const int* __restrict__ bid,
    const unsigned char* __restrict__ vm,
    float* __restrict__ xin, float* __restrict__ xout)
{
  const int bi = blockIdx.x;
  const int b = bi >> 9, i = bi & 511;
  const int t = threadIdx.x, c = t & 31, g = t >> 5;
  __shared__ int sb[ND];
  __shared__ unsigned char sv[ND];
  for (int j = t; j < ND; j += 256) { sb[j] = bid[(b << 9) + j]; sv[j] = vm[(b << 9) + j]; }
  __syncthreads();
  const int bidi = sb[i];
  const bool vmi = sv[i] != 0;
  const bool nmi = bidi >= 0;
  float sin_ = 0.f, sout_ = 0.f, cin_ = 0.f, cout_ = 0.f;
  const float* eb = edge + (long)b * ND * ND * ECH;
  for (int j = g; j < ND; j += 8) {
    int jn = (j + 8 < ND) ? (j + 8) : j;
    __builtin_prefetch(&eb[((long)i * ND + jn) * ECH + c], 0, 3);
    __builtin_prefetch(&eb[((long)jn * ND + i) * ECH + c], 0, 3);
    int bidj = sb[j];
    bool nm  = nmi && (bidj >= 0);
    bool cij = nm && ((bidi > bidj) || ((bidi == bidj) && !vmi));
    bool cji = nm && ((bidj > bidi) || ((bidi == bidj) && (sv[j] == 0)));
    if (cij) { sin_  += eb[((long)i * ND + j) * ECH + c]; cin_  += 1.f; }
    if (cji) { sout_ += eb[((long)j * ND + i) * ECH + c]; cout_ += 1.f; }
  }
  __shared__ float r0[8][32], r1[8][32], r2[8][32], r3[8][32];
  r0[g][c] = sin_; r1[g][c] = sout_; r2[g][c] = cin_; r3[g][c] = cout_;
  __syncthreads();
  if (g == 0) {
    float a = 0.f, bb = 0.f, cc = 0.f, dd = 0.f;
    for (int gg = 0; gg < 8; gg++) { a += r0[gg][c]; bb += r1[gg][c]; cc += r2[gg][c]; dd += r3[gg][c]; }
    xin[(long)bi * ECH + c]  = a  / fmaxf(cc, 1.f);
    xout[(long)bi * ECH + c] = bb / fmaxf(dd, 1.f);
  }
}

// ---------------------------------------------------------------------------
// k_node: h, n1/n2, f1a/f1b, node logits. one block (128 threads) per (b,i)
// ---------------------------------------------------------------------------
__global__ __launch_bounds__(128) void k_node(
    const float* __restrict__ node, const int* __restrict__ bidp,
    const float* __restrict__ xin, const float* __restrict__ xout,
    const float* __restrict__ W_nt, const float* __restrict__ b_nt,
    const float* __restrict__ W_n1, const float* __restrict__ b_n1,
    const float* __restrict__ W_f1, const float* __restrict__ b_f1,
    const float* __restrict__ W_no1, const float* __restrict__ b_no1,
    const float* __restrict__ g_no, const float* __restrict__ be_no,
    const float* __restrict__ W_no2, const float* __restrict__ b_no2,
    float* __restrict__ n1b, float* __restrict__ n2b,
    float* __restrict__ f1ab, float* __restrict__ f1bb,
    float* __restrict__ onode)
{
  const int bi = blockIdx.x;
  const int t = threadIdx.x;
  __shared__ float sin192[192], sh[128], snn[128], sz[128], rbuf[128];
  const float nmf = (bidp[bi] >= 0) ? 1.f : 0.f;
  if (t < 32) { sin192[t] = xin[(long)bi * 32 + t]; sin192[32 + t] = xout[(long)bi * 32 + t]; }
  sin192[64 + t] = node[(long)bi * 128 + t] * nmf;
  __syncthreads();

  float acc = b_nt[t];
  for (int k = 0; k < 192; k++) acc += sin192[k] * W_nt[k * 128 + t];
  float h = silu_f(acc) * nmf;
  sh[t] = h;
  __syncthreads();

  float accn = b_n1[t];
  for (int k = 0; k < 128; k++) accn += sh[k] * W_n1[k * 128 + t];
  snn[t] = accn;
  const long o64 = (long)bi * 64;
  if (t < 64) n1b[o64 + t] = accn; else n2b[o64 + (t - 64)] = accn;
  __syncthreads();

  if (t < 64) {
    float a = 0.f;
    for (int k = 0; k < 64; k++) a += snn[k] * W_f1[k * 64 + t];
    f1ab[o64 + t] = a;
  } else {
    int c = t - 64;
    float a = b_f1[c];
    for (int k = 0; k < 64; k++) a += snn[64 + k] * W_f1[k * 64 + c];
    f1bb[o64 + c] = a;
  }

  float q = b_no1[t];
  for (int k = 0; k < 128; k++) q += sh[k] * W_no1[k * 128 + t];
  rbuf[t] = q; __syncthreads();
  for (int s = 64; s > 0; s >>= 1) { if (t < s) rbuf[t] += rbuf[t + s]; __syncthreads(); }
  float mean = rbuf[0] * (1.f / 128.f);
  __syncthreads();
  float d = q - mean;
  rbuf[t] = d * d; __syncthreads();
  for (int s = 64; s > 0; s >>= 1) { if (t < s) rbuf[t] += rbuf[t + s]; __syncthreads(); }
  float var = rbuf[0] * (1.f / 128.f);
  __syncthreads();
  float y = d * rsqrtf(var + 1e-5f) * g_no[t] + be_no[t];
  sz[t] = silu_f(y);
  __syncthreads();
  if (t < 10) {
    float a = b_no2[t];
    for (int k = 0; k < 128; k++) a += sz[k] * W_no2[k * 10 + t];
    onode[(long)bi * 10 + t] = a * nmf;
  }
}

// ---------------------------------------------------------------------------
// k_edge: fused edge pipeline over one unordered 16x16 tile pair (ti<=tj).
// Phase 1: e_pre for tile + mirror via v_wmma_f32_16x16x32_bf16 (K=32==EC).
// Phase 2: esym from LDS, chained WMMA for W_eo1 (K=64), LN via shuffles,
//          32->5 via shuffles, write (i,j) and (j,i).
// ---------------------------------------------------------------------------
__global__ __launch_bounds__(256) void k_edge(
    const float* __restrict__ edge, const int* __restrict__ bid,
    const unsigned char* __restrict__ vm,
    const float* __restrict__ W_e1, const float* __restrict__ b_e1,
    const float* __restrict__ W2,   const float* __restrict__ b2,
    const float* __restrict__ n1b,  const float* __restrict__ n2b,
    const float* __restrict__ f1ab, const float* __restrict__ f1bb,
    const float* __restrict__ W_eo1, const float* __restrict__ b_eo1,
    const float* __restrict__ g_eo,  const float* __restrict__ be_eo,
    const float* __restrict__ W_eo2, const float* __restrict__ b_eo2,
    float* __restrict__ elog)
{
  const int b = blockIdx.x / NPAIR;
  int p = blockIdx.x % NPAIR;
  int ti = 0;
  { int row = NT; while (p >= row) { p -= row; row--; ti++; } }
  const int tj = ti + p;
  const int i0 = ti * 16, j0 = tj * 16;

  __shared__ float sn1[2][16][64], sn2[2][16][64], sfa[2][16][64], sfb[2][16][64];
  __shared__ __bf16 epre[2][16][16][EPAD];
  __shared__ int   sbid[2][16];
  __shared__ float svmf[2][16];
  __shared__ float sbe1[64], sb2s[64];
  __shared__ float sbeo1[32], sgeo[32], sbeo[32], sW2o[160], sbeo2[8];

  const int t = threadIdx.x;
  for (int x = t; x < 2 * 16 * 64; x += 256) {
    int rs = x >> 10, nn = (x >> 6) & 15, c = x & 63;
    int gn = (rs ? j0 : i0) + nn;
    long off = ((long)(b * ND + gn)) * 64 + c;
    sn1[rs][nn][c] = n1b[off];
    sn2[rs][nn][c] = n2b[off];
    sfa[rs][nn][c] = f1ab[off];
    sfb[rs][nn][c] = f1bb[off];
  }
  if (t < 32) {
    int rs = t >> 4, nn = t & 15;
    int gn = (rs ? j0 : i0) + nn;
    sbid[rs][nn] = bid[b * ND + gn];
    svmf[rs][nn] = (float)vm[b * ND + gn];
  }
  if (t < 64) { sbe1[t] = b_e1[t]; sb2s[t] = b2[t]; }
  if (t < 32) { sbeo1[t] = b_eo1[t]; sgeo[t] = g_eo[t]; sbeo[t] = be_eo[t]; }
  if (t < 160) sW2o[t] = W_eo2[t];
  if (t < 5)   sbeo2[t] = b_eo2[t];

  const int lane = t & 31, wave = t >> 5;
  const int lh = lane >> 4, ln = lane & 15;

  // B registers for GEMM1: Wcat = [W_e1 | W_e1@W_f2], 32x128, bf16 B-layout
  // (elems 0..7 -> K 8*half+e, elems 8..15 -> K 16+8*half+(e-8); lane&15 selects column)
  v16bf B1[8];
#pragma unroll
  for (int nt = 0; nt < 8; nt++) {
    int col = nt * 16 + ln;
#pragma unroll
    for (int e = 0; e < 16; e++) {
      int k = e + ((e >= 8) ? 8 : 0) + 8 * lh;
      float w = (col < 64) ? W_e1[k * 64 + col] : W2[k * 64 + (col - 64)];
      B1[nt][e] = (__bf16)w;
    }
  }
  // B registers for eo1 GEMM (64x32): two K-tiles x two N-tiles
  v16bf Bo[2][2];
#pragma unroll
  for (int kt = 0; kt < 2; kt++)
#pragma unroll
    for (int nt2 = 0; nt2 < 2; nt2++)
#pragma unroll
      for (int e = 0; e < 16; e++)
        Bo[kt][nt2][e] = (__bf16)W_eo1[(kt * 32 + e + ((e >= 8) ? 8 : 0) + 8 * lh) * 32 + (nt2 * 16 + ln)];

  __syncthreads();

  // ---- phase 1: e_pre for tile (tile=0) and mirror (tile=1) ----
  for (int mt = wave; mt < 32; mt += 8) {
    const int tile = mt >> 4, r = mt & 15;
    const int gi = (tile ? j0 : i0) + r;
    const int jbase = tile ? i0 : j0;
    const int c0 = lh * 8;

    // prefetch next M-tile's edge rows (branchless clamp keeps EXEC uniform)
    {
      const int mtn = (mt + 8 < 32) ? (mt + 8) : mt;
      const int tn = mtn >> 4, rn = mtn & 15;
      const int gin = (tn ? j0 : i0) + rn;
      const int jbn = tn ? i0 : j0;
      const float* pref = edge + (((long)b * ND + gin) * ND + (jbn + ln)) * ECH;
      __builtin_prefetch(pref + c0, 0, 3);
      __builtin_prefetch(pref + c0 + 16, 0, 3);
    }

    const int bidi = sbid[tile][r];
    const float vmi = svmf[tile][r];
    const int bidj = sbid[1 - tile][ln];
    const float vmj = svmf[1 - tile][ln];
    const bool nm  = (bidi >= 0) && (bidj >= 0);
    const bool cij = nm && ((bidi > bidj) || ((bidi == bidj) && (vmi == 0.f)));
    const bool cji = nm && ((bidj > bidi) || ((bidi == bidj) && (vmj == 0.f)));
    const float emf = (cij || cji) ? 1.f : 0.f;

    // A: row m = lane&15 -> j = jbase+m; elems 0..7 -> k = c0..c0+7,
    // elems 8..15 -> k = c0+16..c0+23, c0 = 8*(lane>=16)   (ISA A-layout)
    const float* erow = edge + (((long)b * ND + gi) * ND + (jbase + ln)) * ECH;
    float4 a0 = *(const float4*)(erow + c0);
    float4 a1 = *(const float4*)(erow + c0 + 4);
    float4 a2 = *(const float4*)(erow + c0 + 16);
    float4 a3 = *(const float4*)(erow + c0 + 20);
    v16bf A;
    A[0]  = (__bf16)(a0.x * emf); A[1]  = (__bf16)(a0.y * emf);
    A[2]  = (__bf16)(a0.z * emf); A[3]  = (__bf16)(a0.w * emf);
    A[4]  = (__bf16)(a1.x * emf); A[5]  = (__bf16)(a1.y * emf);
    A[6]  = (__bf16)(a1.z * emf); A[7]  = (__bf16)(a1.w * emf);
    A[8]  = (__bf16)(a2.x * emf); A[9]  = (__bf16)(a2.y * emf);
    A[10] = (__bf16)(a2.z * emf); A[11] = (__bf16)(a2.w * emf);
    A[12] = (__bf16)(a3.x * emf); A[13] = (__bf16)(a3.y * emf);
    A[14] = (__bf16)(a3.z * emf); A[15] = (__bf16)(a3.w * emf);

    v8f acc[8];
#pragma unroll
    for (int nt = 0; nt < 8; nt++) {
      v8f z = {};
      acc[nt] = __builtin_amdgcn_wmma_f32_16x16x32_bf16(false, A, false, B1[nt],
                                                        (short)0, z, false, false);
    }

    // epilogue: e_pre = silu(npair + e1 + silu(f1a+f1b)*silu(e2)) -> LDS bf16
#pragma unroll
    for (int nt = 0; nt < 4; nt++) {
      const int c = nt * 16 + ln;
      const float be1v = sbe1[c], b2v = sb2s[c];
      const float n2v = sn2[tile][r][c], fbv = sfb[tile][r][c];
#pragma unroll
      for (int v = 0; v < 8; v++) {
        const int m = v + 8 * lh;               // D-layout row
        float e1 = acc[nt][v] + be1v;
        float e2 = acc[nt + 4][v] + b2v;
        float np = sn1[1 - tile][m][c] + n2v;
        float f1 = silu_f(sfa[1 - tile][m][c] + fbv);
        epre[tile][r][m][c] = (__bf16)silu_f(np + e1 + f1 * silu_f(e2));
      }
    }
  }
  __syncthreads();

  // ---- phase 2: symmetrize + eo pipeline ----
  for (int r2 = wave; r2 < 16; r2 += 8) {
    // esym[dj][k] = epre0[r2][dj][k] + epre1[dj][r2][k]; A rows = dj
    v16bf A0, A1;
#pragma unroll
    for (int e = 0; e < 16; e++) {
      const int k = e + ((e >= 8) ? 8 : 0) + 8 * lh;
      A0[e] = (__bf16)((float)epre[0][r2][ln][k]      + (float)epre[1][ln][r2][k]);
      A1[e] = (__bf16)((float)epre[0][r2][ln][k + 32] + (float)epre[1][ln][r2][k + 32]);
    }
    v8f zz = {};
    v8f q0 = __builtin_amdgcn_wmma_f32_16x16x32_bf16(false, A0, false, Bo[0][0], (short)0, zz, false, false);
    q0     = __builtin_amdgcn_wmma_f32_16x16x32_bf16(false, A1, false, Bo[1][0], (short)0, q0, false, false);
    v8f q1 = __builtin_amdgcn_wmma_f32_16x16x32_bf16(false, A0, false, Bo[0][1], (short)0, zz, false, false);
    q1     = __builtin_amdgcn_wmma_f32_16x16x32_bf16(false, A1, false, Bo[1][1], (short)0, q1, false, false);

    const int bidi = sbid[0][r2];
    const float vmi = svmf[0][r2];
#pragma unroll
    for (int v = 0; v < 8; v++) {
      const int m = v + 8 * lh;          // dj of this output row
      float a  = q0[v] + sbeo1[ln];       // channel ln
      float bq = q1[v] + sbeo1[16 + ln];  // channel 16+ln
      float s = a + bq, ss = a * a + bq * bq;
#pragma unroll
      for (int msk = 1; msk < 16; msk <<= 1) {
        s  += __shfl_xor(s,  msk, 32);
        ss += __shfl_xor(ss, msk, 32);
      }
      float mean = s * (1.f / 32.f);
      float var  = ss * (1.f / 32.f) - mean * mean;
      float rstd = rsqrtf(var + 1e-5f);
      float z0 = silu_f((a  - mean) * rstd * sgeo[ln]      + sbeo[ln]);
      float z1 = silu_f((bq - mean) * rstd * sgeo[16 + ln] + sbeo[16 + ln]);
      float pc[5];
#pragma unroll
      for (int o = 0; o < 5; o++)
        pc[o] = z0 * sW2o[ln * 5 + o] + z1 * sW2o[(16 + ln) * 5 + o];
#pragma unroll
      for (int o = 0; o < 5; o++)
#pragma unroll
        for (int msk = 1; msk < 16; msk <<= 1)
          pc[o] += __shfl_xor(pc[o], msk, 32);

      const int bidj = sbid[1][m];
      const float vmj = svmf[1][m];
      const bool nm  = (bidi >= 0) && (bidj >= 0);
      const bool cij = nm && ((bidi > bidj) || ((bidi == bidj) && (vmi == 0.f)));
      const bool cji = nm && ((bidj > bidi) || ((bidi == bidj) && (vmj == 0.f)));
      const float emf = (cij || cji) ? 1.f : 0.f;
      const long gi_ = i0 + r2, gj_ = j0 + m;
#pragma unroll
      for (int o = 0; o < 5; o++) {
        if (ln == o) {
          float outv = (sbeo2[o] + pc[o]) * emf;
          elog[(((long)b * ND + gi_) * ND + gj_) * 5 + o] = outv;
          elog[(((long)b * ND + gj_) * ND + gi_) * 5 + o] = outv;
        }
      }
    }
  }
}

// ---------------------------------------------------------------------------
extern "C" void kernel_launch(void* const* d_in, const int* in_sizes, int n_in,
                              void* d_out, int out_size, void* d_ws, size_t ws_size,
                              hipStream_t stream) {
  const float* node = (const float*)d_in[0];
  const float* edge = (const float*)d_in[1];
  const int* bid = (const int*)d_in[2];
  const unsigned char* vmm = (const unsigned char*)d_in[3];
  const float* W_nt = (const float*)d_in[4];
  const float* b_nt = (const float*)d_in[5];
  const float* W_n1 = (const float*)d_in[6];
  const float* b_n1 = (const float*)d_in[7];
  const float* W_e1 = (const float*)d_in[8];
  const float* b_e1 = (const float*)d_in[9];
  const float* W_f1 = (const float*)d_in[10];
  const float* b_f1 = (const float*)d_in[11];
  const float* W_f2 = (const float*)d_in[12];
  const float* b_f2 = (const float*)d_in[13];
  const float* W_no1 = (const float*)d_in[14];
  const float* b_no1 = (const float*)d_in[15];
  const float* g_no = (const float*)d_in[16];
  const float* be_no = (const float*)d_in[17];
  const float* W_no2 = (const float*)d_in[18];
  const float* b_no2 = (const float*)d_in[19];
  const float* W_eo1 = (const float*)d_in[20];
  const float* b_eo1 = (const float*)d_in[21];
  const float* g_eo = (const float*)d_in[22];
  const float* be_eo = (const float*)d_in[23];
  const float* W_eo2 = (const float*)d_in[24];
  const float* b_eo2 = (const float*)d_in[25];

  float* ws   = (float*)d_ws;
  float* xin  = ws;                       // 4*512*32
  float* xout = xin  + BD * ND * ECH;     // 4*512*32
  float* n1b  = xout + BD * ND * ECH;     // 4*512*64
  float* n2b  = n1b  + BD * ND * OCH;
  float* f1ab = n2b  + BD * ND * OCH;
  float* f1bb = f1ab + BD * ND * OCH;
  float* W2   = f1bb + BD * ND * OCH;     // 32*64
  float* b2   = W2 + 32 * 64;             // 64

  float* onode = (float*)d_out;
  float* elog  = onode + (long)BD * ND * 10;

  k_prep<<<1, 256, 0, stream>>>(W_e1, b_e1, W_f2, b_f2, W2, b2);
  k_reduce<<<BD * ND, 256, 0, stream>>>(edge, bid, vmm, xin, xout);
  k_node<<<BD * ND, 128, 0, stream>>>(node, bid, xin, xout,
                                      W_nt, b_nt, W_n1, b_n1, W_f1, b_f1,
                                      W_no1, b_no1, g_no, be_no, W_no2, b_no2,
                                      n1b, n2b, f1ab, f1bb, onode);
  k_edge<<<BD * NPAIR, 256, 0, stream>>>(edge, bid, vmm,
                                         W_e1, b_e1, W2, b2,
                                         n1b, n2b, f1ab, f1bb,
                                         W_eo1, b_eo1, g_eo, be_eo, W_eo2, b_eo2,
                                         elog);
}
